// BinaryDense_12111807775177
// MI455X (gfx1250) — compile-verified, matrix operand order checked by reading, not run on hardware
//
#include <hip/hip_runtime.h>

// Problem constants from the reference (fixed shapes).
#define B_SZ   8192
#define D_IN   2048
#define D_OUT  2048

// Tiling: 128x128 output tile per 256-thread block (8 wave32), K-step 32.
#define BM 128
#define BN 128
#define BK 32
// LDS row stride in ushort units: multiple of 8 (keeps ds_load_b128 16B-aligned)
// but not a multiple of 32 -> skews banks between rows.
#define LDSTRIDE 40

typedef __attribute__((ext_vector_type(16))) __bf16       v16bf;
typedef __attribute__((ext_vector_type(8)))  float        v8f;
typedef __attribute__((ext_vector_type(4)))  unsigned int v4u;

union Frag {
    v16bf v;
    v4u   q[2];   // two 16-byte chunks per 16x32 bf16 fragment
};

// float -> bf16 (round-to-nearest-even), bit-level.
__device__ __forceinline__ unsigned short f2bf_rne(float f) {
    unsigned int u = __float_as_uint(f);
    u += 0x7FFFu + ((u >> 16) & 1u);
    return (unsigned short)(u >> 16);
}

// =====================================================================
// Pre-pass 1: split X (f32) into bf16 hi/lo planes (x = hi + lo).
// =====================================================================
__global__ __launch_bounds__(256)
void split_x_kernel(const float* __restrict__ X,
                    unsigned short* __restrict__ Xhi,
                    unsigned short* __restrict__ Xlo) {
    const size_t i = ((size_t)blockIdx.x * 256 + threadIdx.x) * 4;
    const float4 v = *(const float4*)(X + i);
    float f[4] = {v.x, v.y, v.z, v.w};
    unsigned int hp[2], lp[2];
    #pragma unroll
    for (int e = 0; e < 2; ++e) {
        const unsigned short ha = f2bf_rne(f[2 * e + 0]);
        const unsigned short hb = f2bf_rne(f[2 * e + 1]);
        const float ra = f[2 * e + 0] - __uint_as_float((unsigned int)ha << 16);
        const float rb = f[2 * e + 1] - __uint_as_float((unsigned int)hb << 16);
        hp[e] = (unsigned int)ha | ((unsigned int)hb << 16);
        lp[e] = (unsigned int)f2bf_rne(ra) | ((unsigned int)f2bf_rne(rb) << 16);
    }
    ((unsigned int*)(Xhi + i))[0] = hp[0];
    ((unsigned int*)(Xhi + i))[1] = hp[1];
    ((unsigned int*)(Xlo + i))[0] = lp[0];
    ((unsigned int*)(Xlo + i))[1] = lp[1];
}

// =====================================================================
// Pre-pass 2: binarize W ([k][n] f32) and transpose to Wt ([n][k] bf16).
// 64x64 tiles through LDS, 256 threads.
// =====================================================================
#define TW_STRIDE 72   // ushort; multiple of 8, bank-skewed
__global__ __launch_bounds__(256)
void binarize_transpose_w_kernel(const float* __restrict__ W,
                                 unsigned short* __restrict__ Wt) {
    __shared__ __align__(16) unsigned short t[64 * TW_STRIDE];
    const int tid = threadIdx.x;
    const int bx = blockIdx.x * 64;   // n base
    const int by = blockIdx.y * 64;   // k base

    // Load 64(k) x 64(n) f32, binarize, store transposed into LDS t[n][k].
    const int c4 = (tid & 15) << 2;   // n-in-tile: 0..60
    const int r0 = tid >> 4;          // k-in-tile base: 0..15
    #pragma unroll
    for (int rr = 0; rr < 64; rr += 16) {
        const int r = r0 + rr;
        const float4 v = *(const float4*)(W + (size_t)(by + r) * D_OUT + bx + c4);
        t[(c4 + 0) * TW_STRIDE + r] = (v.x < 0.f) ? (unsigned short)0u : (unsigned short)0x3F80u;
        t[(c4 + 1) * TW_STRIDE + r] = (v.y < 0.f) ? (unsigned short)0u : (unsigned short)0x3F80u;
        t[(c4 + 2) * TW_STRIDE + r] = (v.z < 0.f) ? (unsigned short)0u : (unsigned short)0x3F80u;
        t[(c4 + 3) * TW_STRIDE + r] = (v.w < 0.f) ? (unsigned short)0u : (unsigned short)0x3F80u;
    }
    __syncthreads();

    // Write coalesced rows of Wt: each thread writes 16 bf16 of one n-row.
    const int n0 = tid >> 2;          // 0..63
    const int k8 = (tid & 3) << 4;    // 0,16,32,48
    unsigned short* dst = Wt + (size_t)(bx + n0) * D_IN + by + k8;
    *(v4u*)(dst)     = *(const v4u*)&t[n0 * TW_STRIDE + k8];
    *(v4u*)(dst + 8) = *(const v4u*)&t[n0 * TW_STRIDE + k8 + 8];
}

// =====================================================================
// Main GEMM (pre-converted bf16 operands): zero in-loop conversion VALU,
// ping-pong LDS double buffering, one barrier per K-step.
// =====================================================================
__global__ __launch_bounds__(256)
void gemm_bf16_wmma(const unsigned short* __restrict__ Xhi,
                    const unsigned short* __restrict__ Xlo,
                    const unsigned short* __restrict__ Wt,
                    float* __restrict__ Out) {
    __shared__ __align__(16) unsigned short sAhi[2][BM * LDSTRIDE];
    __shared__ __align__(16) unsigned short sAlo[2][BM * LDSTRIDE];
    __shared__ __align__(16) unsigned short sBt [2][BN * LDSTRIDE];

    const int tid   = threadIdx.x;
    const int lane  = tid & 31;
    const int wave  = tid >> 5;
    const int waveM = wave >> 2;          // 0..1 -> 64 rows each
    const int waveN = wave & 3;           // 0..3 -> 32 cols each
    const int l15   = lane & 15;
    const int kb    = (lane >> 4) << 3;   // K base: 0 or 8

    const int rowBase = blockIdx.y * BM;
    const int colBase = blockIdx.x * BN;

    // Staging map: 512 b128 chunks per tile -> 2 per thread (rows r, r+64).
    const int ldr = tid >> 2;             // 0..63
    const int ldq = (tid & 3) << 3;       // bf16 col offset: 0,8,16,24
    const int sa  = ldr * LDSTRIDE + ldq;
    const int sa2 = (ldr + 64) * LDSTRIDE + ldq;

    const unsigned short* pAhi0 = Xhi + (size_t)(rowBase + ldr)      * D_IN + ldq;
    const unsigned short* pAhi1 = Xhi + (size_t)(rowBase + ldr + 64) * D_IN + ldq;
    const unsigned short* pAlo0 = Xlo + (size_t)(rowBase + ldr)      * D_IN + ldq;
    const unsigned short* pAlo1 = Xlo + (size_t)(rowBase + ldr + 64) * D_IN + ldq;
    const unsigned short* pB0   = Wt  + (size_t)(colBase + ldr)      * D_IN + ldq;
    const unsigned short* pB1   = Wt  + (size_t)(colBase + ldr + 64) * D_IN + ldq;

    v8f acc[4][2] = {};

    // Prologue: stage tile kt=0 into buffer 0.
    *(v4u*)&sAhi[0][sa]  = *(const v4u*)(pAhi0);
    *(v4u*)&sAhi[0][sa2] = *(const v4u*)(pAhi1);
    *(v4u*)&sAlo[0][sa]  = *(const v4u*)(pAlo0);
    *(v4u*)&sAlo[0][sa2] = *(const v4u*)(pAlo1);
    *(v4u*)&sBt [0][sa]  = *(const v4u*)(pB0);
    *(v4u*)&sBt [0][sa2] = *(const v4u*)(pB1);
    __syncthreads();

    int p = 0;
    for (int kt = 0; kt < D_IN; kt += BK) {
        const bool hasNext = (kt + BK) < D_IN;

        // Issue next tile's global loads early (overlap with WMMAs below).
        v4u nAhi0, nAhi1, nAlo0, nAlo1, nB0, nB1;
        if (hasNext) {
            const int ko = kt + BK;
            nAhi0 = *(const v4u*)(pAhi0 + ko);
            nAhi1 = *(const v4u*)(pAhi1 + ko);
            nAlo0 = *(const v4u*)(pAlo0 + ko);
            nAlo1 = *(const v4u*)(pAlo1 + ko);
            nB0   = *(const v4u*)(pB0   + ko);
            nB1   = *(const v4u*)(pB1   + ko);
        }
        // L2 prefetch two tiles ahead (global_prefetch_b8).
        if (kt + 2 * BK < D_IN) {
            __builtin_prefetch(pAhi0 + kt + 2 * BK, 0, 1);
            __builtin_prefetch(pB0   + kt + 2 * BK, 0, 1);
        }

        // Compute current tile from buffer p: 16 WMMAs (hi + lo passes).
        Frag bfr[2];
        #pragma unroll
        for (int ni = 0; ni < 2; ++ni) {
            const int c = waveN * 32 + ni * 16 + l15;
            const int a = c * LDSTRIDE + kb;
            bfr[ni].q[0] = *(const v4u*)&sBt[p][a];
            bfr[ni].q[1] = *(const v4u*)&sBt[p][a + 16];
        }
        #pragma unroll
        for (int mi = 0; mi < 4; ++mi) {
            const int r = waveM * 64 + mi * 16 + l15;
            const int a = r * LDSTRIDE + kb;
            Frag fh, fl;
            fh.q[0] = *(const v4u*)&sAhi[p][a];
            fh.q[1] = *(const v4u*)&sAhi[p][a + 16];
            fl.q[0] = *(const v4u*)&sAlo[p][a];
            fl.q[1] = *(const v4u*)&sAlo[p][a + 16];
            #pragma unroll
            for (int ni = 0; ni < 2; ++ni) {
                acc[mi][ni] = __builtin_amdgcn_wmma_f32_16x16x32_bf16(
                    false, fh.v, false, bfr[ni].v, (short)0, acc[mi][ni], false, false);
                acc[mi][ni] = __builtin_amdgcn_wmma_f32_16x16x32_bf16(
                    false, fl.v, false, bfr[ni].v, (short)0, acc[mi][ni], false, false);
            }
        }

        // Store next tile into the other buffer; single barrier per K-step.
        if (hasNext) {
            const int pn = p ^ 1;
            *(v4u*)&sAhi[pn][sa]  = nAhi0;
            *(v4u*)&sAhi[pn][sa2] = nAhi1;
            *(v4u*)&sAlo[pn][sa]  = nAlo0;
            *(v4u*)&sAlo[pn][sa2] = nAlo1;
            *(v4u*)&sBt [pn][sa]  = nB0;
            *(v4u*)&sBt [pn][sa2] = nB1;
        }
        __syncthreads();
        p ^= 1;
    }

    // Epilogue: C/D layout -> global. VGPR r holds M = r + 8*(lane>=16), N = lane&15.
    const int rbase = rowBase + waveM * 64 + ((lane >> 4) << 3);
    #pragma unroll
    for (int mi = 0; mi < 4; ++mi) {
        #pragma unroll
        for (int ni = 0; ni < 2; ++ni) {
            const int col = colBase + waveN * 32 + ni * 16 + l15;
            #pragma unroll
            for (int r = 0; r < 8; ++r) {
                Out[(size_t)(rbase + mi * 16 + r) * D_OUT + col] = acc[mi][ni][r];
            }
        }
    }
}

// =====================================================================
// Fallback: fused kernel (no workspace needed) — from round 1.
// =====================================================================
__global__ __launch_bounds__(256)
void binary_dense_wmma(const float* __restrict__ X,
                       const float* __restrict__ W,
                       float* __restrict__ Out) {
    __shared__ __align__(16) unsigned short Ahi[BM * LDSTRIDE];
    __shared__ __align__(16) unsigned short Alo[BM * LDSTRIDE];
    __shared__ __align__(16) unsigned short Bs [BN * LDSTRIDE];

    const int tid   = threadIdx.x;
    const int lane  = tid & 31;
    const int wave  = tid >> 5;
    const int waveM = wave >> 2;
    const int waveN = wave & 3;
    const int l15   = lane & 15;
    const int kb    = (lane >> 4) << 3;

    const int rowBase = blockIdx.y * BM;
    const int colBase = blockIdx.x * BN;

    v8f acc[4][2] = {};

    for (int kt = 0; kt < D_IN; kt += BK) {
        #pragma unroll
        for (int i = 0; i < 4; ++i) {
            const int idx = tid + i * 256;
            {
                const int r  = idx >> 3;
                const int c4 = (idx & 7) << 2;
                const float4 v = *(const float4*)(X + (size_t)(rowBase + r) * D_IN + kt + c4);
                float f[4] = {v.x, v.y, v.z, v.w};
                unsigned int hp[2], lp[2];
                #pragma unroll
                for (int e = 0; e < 2; ++e) {
                    const unsigned short ha = f2bf_rne(f[2 * e + 0]);
                    const unsigned short hb = f2bf_rne(f[2 * e + 1]);
                    const float ra = f[2 * e + 0] - __uint_as_float((unsigned int)ha << 16);
                    const float rb = f[2 * e + 1] - __uint_as_float((unsigned int)hb << 16);
                    hp[e] = (unsigned int)ha | ((unsigned int)hb << 16);
                    lp[e] = (unsigned int)f2bf_rne(ra) | ((unsigned int)f2bf_rne(rb) << 16);
                }
                const int a = r * LDSTRIDE + c4;
                ((unsigned int*)&Ahi[a])[0] = hp[0];
                ((unsigned int*)&Ahi[a])[1] = hp[1];
                ((unsigned int*)&Alo[a])[0] = lp[0];
                ((unsigned int*)&Alo[a])[1] = lp[1];
            }
            {
                const int kr = idx >> 5;
                const int c4 = (idx & 31) << 2;
                const float4 v = *(const float4*)(W + (size_t)(kt + kr) * D_OUT + colBase + c4);
                Bs[(c4 + 0) * LDSTRIDE + kr] = (v.x < 0.f) ? (unsigned short)0u : (unsigned short)0x3F80u;
                Bs[(c4 + 1) * LDSTRIDE + kr] = (v.y < 0.f) ? (unsigned short)0u : (unsigned short)0x3F80u;
                Bs[(c4 + 2) * LDSTRIDE + kr] = (v.z < 0.f) ? (unsigned short)0u : (unsigned short)0x3F80u;
                Bs[(c4 + 3) * LDSTRIDE + kr] = (v.w < 0.f) ? (unsigned short)0u : (unsigned short)0x3F80u;
            }
        }
        __syncthreads();

        Frag bfr[2];
        #pragma unroll
        for (int ni = 0; ni < 2; ++ni) {
            const int c = waveN * 32 + ni * 16 + l15;
            const int a = c * LDSTRIDE + kb;
            bfr[ni].q[0] = *(const v4u*)&Bs[a];
            bfr[ni].q[1] = *(const v4u*)&Bs[a + 16];
        }
        #pragma unroll
        for (int mi = 0; mi < 4; ++mi) {
            const int r = waveM * 64 + mi * 16 + l15;
            const int a = r * LDSTRIDE + kb;
            Frag fh, fl;
            fh.q[0] = *(const v4u*)&Ahi[a];
            fh.q[1] = *(const v4u*)&Ahi[a + 16];
            fl.q[0] = *(const v4u*)&Alo[a];
            fl.q[1] = *(const v4u*)&Alo[a + 16];
            #pragma unroll
            for (int ni = 0; ni < 2; ++ni) {
                acc[mi][ni] = __builtin_amdgcn_wmma_f32_16x16x32_bf16(
                    false, fh.v, false, bfr[ni].v, (short)0, acc[mi][ni], false, false);
                acc[mi][ni] = __builtin_amdgcn_wmma_f32_16x16x32_bf16(
                    false, fl.v, false, bfr[ni].v, (short)0, acc[mi][ni], false, false);
            }
        }
        __syncthreads();
    }

    const int rbase = rowBase + waveM * 64 + ((lane >> 4) << 3);
    #pragma unroll
    for (int mi = 0; mi < 4; ++mi) {
        #pragma unroll
        for (int ni = 0; ni < 2; ++ni) {
            const int col = colBase + waveN * 32 + ni * 16 + l15;
            #pragma unroll
            for (int r = 0; r < 8; ++r) {
                Out[(size_t)(rbase + mi * 16 + r) * D_OUT + col] = acc[mi][ni][r];
            }
        }
    }
}

extern "C" void kernel_launch(void* const* d_in, const int* in_sizes, int n_in,
                              void* d_out, int out_size, void* d_ws, size_t ws_size,
                              hipStream_t stream) {
    (void)in_sizes; (void)n_in; (void)out_size;
    const float* X = (const float*)d_in[0];   // [8192, 2048] f32
    const float* W = (const float*)d_in[1];   // [2048, 2048] f32
    float* Out = (float*)d_out;               // [8192, 2048] f32

    const size_t nX = (size_t)B_SZ * D_IN;    // 16.7M elems
    const size_t nW = (size_t)D_IN * D_OUT;   // 4.2M elems
    const size_t need = (2 * nX + nW) * sizeof(unsigned short);  // 72 MB

    dim3 grid(D_OUT / BN, B_SZ / BM);         // (16, 64)

    if (d_ws != nullptr && ws_size >= need) {
        unsigned short* Xhi = (unsigned short*)d_ws;
        unsigned short* Xlo = Xhi + nX;
        unsigned short* Wt  = Xlo + nX;

        split_x_kernel<<<(unsigned)(nX / (256 * 4)), 256, 0, stream>>>(X, Xhi, Xlo);
        binarize_transpose_w_kernel<<<dim3(D_OUT / 64, D_IN / 64), 256, 0, stream>>>(W, Wt);
        gemm_bf16_wmma<<<grid, 256, 0, stream>>>(Xhi, Xlo, Wt, Out);
    } else {
        binary_dense_wmma<<<grid, 256, 0, stream>>>(X, W, Out);
    }
}
